// ADKGenerator_21449066676363
// MI455X (gfx1250) — matrix operand order repaired and checked by reading, not compiled, not run on hardware
//
#include <hip/hip_runtime.h>
#include <math.h>
#include <stdint.h>

#define B_   8
#define C_   32
#define H_   256
#define W_   256
#define NK   9            // K*K
#define HW_  (H_ * W_)

#if defined(__has_builtin)
#if __has_builtin(__builtin_amdgcn_tensor_load_to_lds)
#define HAVE_TDM 1
#endif
#endif
#ifndef HAVE_TDM
#define HAVE_TDM 0
#endif

typedef __attribute__((ext_vector_type(16))) _Float16     v16h;
typedef __attribute__((ext_vector_type(8)))  _Float16     v8h;
typedef __attribute__((ext_vector_type(8)))  float        v8f;
typedef __attribute__((ext_vector_type(4)))  unsigned int v4u;
typedef __attribute__((ext_vector_type(8)))  int          v8i;
typedef __attribute__((ext_vector_type(4)))  int          v4i;

__device__ __forceinline__ float leaky2(float v) { return v >= 0.f ? v : 0.2f * v; }

// A-fragment K index for the 16-bit 16x32 A layout (lane l, element e):
//   lanes 0-15: V0..V3 -> K 0..7, V4..V7 -> K 16..23 ; lanes 16-31: +8
__device__ __forceinline__ int a_kloc(int e, int halfsel) {
    const int pe = e >> 1;
    return (pe < 4 ? pe * 2 : 16 + (pe - 4) * 2) + halfsel * 8 + (e & 1);
}

// ---------------------------------------------------------------------------
// Kernel 1: GAP of x over HxW per (b,c). Streams 64 MB with prefetch-ahead.
// ---------------------------------------------------------------------------
__global__ void k_gap(const float* __restrict__ x, float* __restrict__ gap) {
    __shared__ float red[256];
    const int bc = blockIdx.x;
    const float* p = x + (size_t)bc * HW_;
    float s = 0.f;
    for (int i = threadIdx.x; i < HW_; i += 256) {
        __builtin_prefetch(p + ((i + 4096) & (HW_ - 1)), 0, 0);
        s += p[i];
    }
    red[threadIdx.x] = s;
    __syncthreads();
    for (int off = 128; off > 0; off >>= 1) {
        if (threadIdx.x < (unsigned)off) red[threadIdx.x] += red[threadIdx.x + off];
        __syncthreads();
    }
    if (threadIdx.x == 0) gap[bc] = red[0] * (1.0f / HW_);
}

// ---------------------------------------------------------------------------
// Kernel 2: spectral branch -> spec[b,c,9]. One block; tid -> (b,c).
// ---------------------------------------------------------------------------
__global__ void k_spec(const float* __restrict__ gap,
                       const float* __restrict__ w_s1, const float* __restrict__ b_s1,
                       const float* __restrict__ w_s2, const float* __restrict__ b_s2,
                       const float* __restrict__ norm_std,
                       float* __restrict__ spec) {
    __shared__ float part[256];
    __shared__ float g1[B_];
    const int tid = threadIdx.x;
    const int b = tid >> 5, c = tid & 31;
    part[tid] = w_s1[c] * gap[b * C_ + c];
    __syncthreads();
    if (c == 0) {
        float s = b_s1[0];
        for (int i = 0; i < C_; ++i) s += part[b * C_ + i];
        g1[b] = leaky2(s);
    }
    __syncthreads();
    float g[NK];
    float m = 0.f;
    for (int t = 0; t < NK; ++t) {
        g[t] = w_s2[c * NK + t] * g1[b] + b_s2[c * NK + t];
        m += g[t];
    }
    m *= (1.f / NK);
    float var = 0.f;
    for (int t = 0; t < NK; ++t) { float d = g[t] - m; var += d * d; }
    var *= (1.f / (NK - 1));
    const float sd = sqrtf(var);
    for (int t = 0; t < NK; ++t)
        spec[(size_t)(b * C_ + c) * NK + t] = (g[t] - m) / (sd + 1e-10f) * norm_std[c * NK + t];
}

// ---------------------------------------------------------------------------
// Kernel 3: fused depthwise 3x3 + pointwise 32->9, one WMMA per 16 px/wave.
// t1 stored f16, pixel-major, padded to 16 channels (pad channels == 0):
// t1p[(b*HW + p)*16 + j] -> one 16-B store per lane.
// ---------------------------------------------------------------------------
__global__ void k_dwpw(const float* __restrict__ y,
                       const float* __restrict__ w_dw, const float* __restrict__ b_dw,
                       const float* __restrict__ w_p1, const float* __restrict__ b_p1,
                       _Float16* __restrict__ t1p) {
    __shared__ float s_wdw[C_ * 9];
    __shared__ float s_bdw[C_];
    __shared__ float s_bp1[16];
    __shared__ __align__(32) _Float16 s_afrag[32][16];
    const int tid = threadIdx.x;

    for (int i = tid; i < C_ * 9; i += 256) s_wdw[i] = w_dw[i];
    if (tid < C_) s_bdw[tid] = b_dw[tid];
    if (tid < 16) s_bp1[tid] = (tid < NK) ? b_p1[tid] : 0.f;
    for (int i = tid; i < 512; i += 256) {
        const int l = i >> 4, e = i & 15;
        const int M = l & 15, hf = l >> 4;
        const int k = a_kloc(e, hf);
        s_afrag[l][e] = (M < NK) ? (_Float16)w_p1[M * C_ + k] : (_Float16)0.f;
    }
    __syncthreads();

    const int blocksPerBatch = HW_ / 128;
    const int b   = blockIdx.x / blocksPerBatch;
    const int seg = blockIdx.x % blocksPerBatch;
    const int wv = tid >> 5, lane = tid & 31;
    const int n = lane & 15, half = lane >> 4;
    const int p = seg * 128 + wv * 16 + n;
    const int h = p >> 8, w = p & (W_ - 1);

    float dwv[16];
    #pragma unroll
    for (int cc = 0; cc < 16; ++cc) {
        const int c = (cc < 8) ? (8 * half + cc) : (16 + 8 * half + (cc - 8));
        const float* yc = y + (size_t)(b * C_ + c) * HW_;
        float acc = s_bdw[c];
        #pragma unroll
        for (int di = 0; di < 3; ++di) {
            const int hy = h + di - 1;
            const float mh = (hy >= 0 && hy < H_) ? 1.f : 0.f;
            const int hyc = hy < 0 ? 0 : (hy > H_ - 1 ? H_ - 1 : hy);
            #pragma unroll
            for (int dj = 0; dj < 3; ++dj) {
                const int wy = w + dj - 1;
                const float mw = (wy >= 0 && wy < W_) ? 1.f : 0.f;
                const int wyc = wy < 0 ? 0 : (wy > W_ - 1 ? W_ - 1 : wy);
                acc += (s_wdw[c * 9 + di * 3 + dj] * (mh * mw)) * yc[hyc * W_ + wyc];
            }
        }
        dwv[cc] = acc;
    }

    v16h bm;
    #pragma unroll
    for (int e = 0; e < 16; ++e) bm[e] = (_Float16)dwv[e];
    const v16h am = *(const v16h*)(&s_afrag[lane][0]);

    v8f acc = {};
    acc = __builtin_amdgcn_wmma_f32_16x16x32_f16(false, am, false, bm,
                                                 (short)0, acc, false, false);

    v8h outv;
    #pragma unroll
    for (int r = 0; r < 8; ++r)
        outv[r] = (_Float16)(acc[r] + s_bp1[r + 8 * half]);
    *(v8h*)(t1p + ((size_t)b * HW_ + p) * 16 + 8 * half) = outv;
}

// ---------------------------------------------------------------------------
// Kernel 4: conv2+leaky then conv3+channel-norm as tap-pair WMMA GEMMs.
// One 3x3 tap = one contiguous 16-ch (32 B) record = half a K=32 WMMA:
//   K 0-15  = record at tap 2j   (lanes 0-15)
//   K 16-31 = record at tap 2j+1 (lanes 16-31)
// -> B fragment is ONE 32-B LDS vector load per lane; 5 WMMAs per conv.
// t1 halo tile loaded by the Tensor Data Mover for interior tiles
// (2-D D#, 20 rows x 640 B, stride 8192 B), async->LDS b128 otherwise.
// ---------------------------------------------------------------------------
__global__ void k_conv23(const _Float16* __restrict__ t1p,
                         const float* __restrict__ w_c2, const float* __restrict__ b_c2,
                         const float* __restrict__ w_c3, const float* __restrict__ b_c3,
                         float std_spatial,
                         float* __restrict__ spat) {
    __shared__ __align__(32) _Float16 s_t1[400][16];     // 20x20 halo-2 tile
    __shared__ __align__(32) _Float16 s_t2[324][16];     // 18x18 halo-1 tile
    __shared__ __align__(32) _Float16 s_a2[5][32][16];   // A fragments conv2
    __shared__ __align__(32) _Float16 s_a3[5][32][16];   // A fragments conv3
    __shared__ __align__(16) float    s_v[256][16];      // conv3 result, padded
    __shared__ float s_b2[16], s_b3[16];

    const int b  = blockIdx.z;
    const int h0 = blockIdx.y * 16, w0 = blockIdx.x * 16;
    const int tid = threadIdx.x;
    const int wv = tid >> 5, lane = tid & 31, n = lane & 15, half = lane >> 4;

    if (tid < 16) {
        s_b2[tid] = (tid < NK) ? b_c2[tid] : 0.f;
        s_b3[tid] = (tid < NK) ? b_c3[tid] : 0.f;
    }
    // A fragments: 5 tap-pairs x 32 lanes x 16 elems, for both convs.
    // k<16 -> (tap=2*tp, ic=k) ; k>=16 -> (tap=2*tp+1, ic=k-16)
    for (int i = tid; i < 5 * 512; i += 256) {
        const int tp = i >> 9, rem = i & 511, l = rem >> 4, e = rem & 15;
        const int M = l & 15, hf = l >> 4;
        const int k = hf * 16 + e;
        const int tap = 2 * tp + (k >> 4);
        const int ic  = k & 15;
        _Float16 a2 = (_Float16)0.f, a3 = (_Float16)0.f;
        if (M < NK && ic < NK && tap < NK) {             // w: [oc*81 + ic*9 + tap]
            a2 = (_Float16)w_c2[M * 81 + ic * NK + tap];
            a3 = (_Float16)w_c3[M * 81 + ic * NK + tap];
        }
        s_a2[tp][l][e] = a2;
        s_a3[tp][l][e] = a3;
    }

    // ---- t1 tile fill: TDM for interior tiles, async->LDS b128 for borders
    const bool interior = HAVE_TDM &&
        (h0 >= 2) && (h0 + 18 <= H_) && (w0 >= 2) && (w0 + 18 <= W_);
    if (interior) {
#if HAVE_TDM
        if (wv == 0) {
            const uint64_t ga =
                (uint64_t)(uintptr_t)(t1p + ((size_t)b * HW_ + (h0 - 2) * W_ + (w0 - 2)) * 16);
            const unsigned lds = (uint32_t)(uintptr_t)(&s_t1[0][0]);
            v4u g0;                                      // D# group 0 (ISA 8.3)
            g0[0] = 1u;                                  // count=1, user descriptor
            g0[1] = lds;                                 // lds_addr
            g0[2] = (unsigned)ga;                        // global_addr[31:0]
            g0[3] = ((unsigned)(ga >> 32) & 0x1FFFFFFu)  // global_addr[56:32]
                    | 0x80000000u;                       // type=2 ("image")
            v8i g1;                                      // D# group 1 (ISA 8.4)
            g1[0] = 0x00010000;                          // data_size=1 (2 bytes)
            g1[1] = (int)((unsigned)(W_ * 16) << 16);    // tensor_dim0[15:0]=4096
            g1[2] = (int)((unsigned)H_ << 16);           // tensor_dim1[15:0]=256
            g1[3] = (int)(320u << 16);                   // tile_dim0 = 20*16
            g1[4] = 20;                                  // tile_dim1 = 20 rows
            g1[5] = W_ * 16;                             // tensor_dim0_stride=4096
            g1[6] = 0;
            g1[7] = 0;
            const v4i gz4 = {0, 0, 0, 0};                // 2-D: groups 2/3 unused
            const v8i gz8 = {0, 0, 0, 0, 0, 0, 0, 0};
            __builtin_amdgcn_tensor_load_to_lds(g0, g1, gz4, gz4, gz8, 0);
            __builtin_amdgcn_s_wait_tensorcnt((short)0);
        }
#endif
    } else {
        for (int i = tid; i < 800; i += 256) {
            const int pix = i >> 1, part = i & 1;
            const int lh = pix / 20, lw = pix % 20;
            const int gh = h0 - 2 + lh, gw = w0 - 2 + lw;
            _Float16* dst = &s_t1[pix][part * 8];
            if ((unsigned)gh < (unsigned)H_ && (unsigned)gw < (unsigned)W_) {
                const _Float16* src =
                    t1p + ((size_t)b * HW_ + gh * W_ + gw) * 16 + part * 8;
                const uint32_t lds = (uint32_t)(uintptr_t)dst;
                const uint64_t ga  = (uint64_t)(uintptr_t)src;
                asm volatile("global_load_async_to_lds_b128 %0, %1, off"
                             :: "v"(lds), "v"(ga) : "memory");
            } else {
                *(float4*)dst = make_float4(0.f, 0.f, 0.f, 0.f);
            }
        }
        asm volatile("s_wait_asynccnt 0x0" ::: "memory");
    }
    __syncthreads();

    // ---- conv2 + leaky over the 18x18 halo-1 grid (21 pixel-groups) ----
    for (int g = wv; g < 21; g += 8) {
        int p = g * 16 + n; if (p > 323) p = 323;        // clamp (dup writes ok)
        const int lh = p / 18, lw = p % 18;
        v8f acc = {};
        #pragma unroll
        for (int tp = 0; tp < 5; ++tp) {
            const v16h a = *(const v16h*)(&s_a2[tp][lane][0]);
            int tap = 2 * tp + half; if (tap > 8) tap = 8;   // A half is 0 there
            const v16h bf =
                *(const v16h*)(&s_t1[(lh + tap / 3) * 20 + (lw + tap % 3)][0]);
            acc = __builtin_amdgcn_wmma_f32_16x16x32_f16(false, a, false, bf,
                                                         (short)0, acc, false, false);
        }
        v8h r2;
        #pragma unroll
        for (int r = 0; r < 8; ++r)
            r2[r] = (_Float16)leaky2(acc[r] + s_b2[r + 8 * half]);
        *(v8h*)(&s_t2[p][8 * half]) = r2;
    }
    __syncthreads();

    // ---- conv3 over the 16x16 output tile (16 pixel-groups) ----
    for (int g = wv; g < 16; g += 8) {
        const int p = g * 16 + n;
        const int lh = p >> 4, lw = p & 15;
        v8f acc = {};
        #pragma unroll
        for (int tp = 0; tp < 5; ++tp) {
            const v16h a = *(const v16h*)(&s_a3[tp][lane][0]);
            int tap = 2 * tp + half; if (tap > 8) tap = 8;
            const v16h bf =
                *(const v16h*)(&s_t2[(lh + tap / 3) * 18 + (lw + tap % 3)][0]);
            acc = __builtin_amdgcn_wmma_f32_16x16x32_f16(false, a, false, bf,
                                                         (short)0, acc, false, false);
        }
        *(float4*)(&s_v[p][8 * half])     = make_float4(acc[0] + s_b3[8 * half + 0],
                                                        acc[1] + s_b3[8 * half + 1],
                                                        acc[2] + s_b3[8 * half + 2],
                                                        acc[3] + s_b3[8 * half + 3]);
        *(float4*)(&s_v[p][8 * half + 4]) = make_float4(acc[4] + s_b3[8 * half + 4],
                                                        acc[5] + s_b3[8 * half + 5],
                                                        acc[6] + s_b3[8 * half + 6],
                                                        acc[7] + s_b3[8 * half + 7]);
    }
    __syncthreads();

    // ---- per-pixel norm over the 9 channels (ddof=1) * STD_SPATIAL ----
    {
        const int p = tid, ty = p >> 4, tx = p & 15;
        float v[NK], m = 0.f;
        #pragma unroll
        for (int t = 0; t < NK; ++t) { v[t] = s_v[p][t]; m += v[t]; }
        m *= (1.f / NK);
        float var = 0.f;
        #pragma unroll
        for (int t = 0; t < NK; ++t) { float d = v[t] - m; var += d * d; }
        var *= (1.f / (NK - 1));
        const float sd = sqrtf(var);
        const size_t base = (((size_t)b * H_ + h0 + ty) * W_ + w0 + tx) * NK;
        #pragma unroll
        for (int t = 0; t < NK; ++t)
            spat[base + t] = (v[t] - m) / (sd + 1e-10f) * std_spatial;
    }
}

// ---------------------------------------------------------------------------
// Kernel 5: AC4IF gather. 16x16 pixel tile, all 32 channels. x tile loaded
// with aligned async b128 chunks into a 24-float padded row ([w0-4, w0+20)).
// ---------------------------------------------------------------------------
__global__ void k_out(const float* __restrict__ x,
                      const float* __restrict__ spec,
                      const float* __restrict__ spat,
                      float* __restrict__ out) {
    __shared__ __align__(16) float s_x[C_][18][24];      // 55.3 KB
    __shared__ float s_spec[C_][NK];
    const int b  = blockIdx.z;
    const int h0 = blockIdx.y * 16, w0 = blockIdx.x * 16;
    const int tid = threadIdx.x;

    for (int i = tid; i < C_ * NK; i += 256)
        s_spec[i / NK][i % NK] = spec[(size_t)b * C_ * NK + i];
    // 6 aligned float4 chunks per row: columns [w0-4, w0+20)
    for (int i = tid; i < C_ * 18 * 6; i += 256) {
        const int c = i / 108, rem = i % 108, lh = rem / 6, ck = rem % 6;
        const int gh = h0 - 1 + lh;
        const int gws = w0 - 4 + ck * 4;
        float* dst = &s_x[c][lh][ck * 4];
        if ((unsigned)gh < (unsigned)H_ && gws >= 0 && gws + 4 <= W_) {
            const float* src = x + (size_t)(b * C_ + c) * HW_ + gh * W_ + gws;
            const uint32_t lds = (uint32_t)(uintptr_t)dst;
            const uint64_t ga  = (uint64_t)(uintptr_t)src;
            asm volatile("global_load_async_to_lds_b128 %0, %1, off"
                         :: "v"(lds), "v"(ga) : "memory");
        } else {
            *(float4*)dst = make_float4(0.f, 0.f, 0.f, 0.f);
        }
    }
    asm volatile("s_wait_asynccnt 0x0" ::: "memory");
    __syncthreads();

    const int ty = tid / 16, tx = tid % 16;
    const int h = h0 + ty, w = w0 + tx;
    float sp[NK];
    const size_t sbase = (((size_t)b * H_ + h) * W_ + w) * NK;
    #pragma unroll
    for (int t = 0; t < NK; ++t) sp[t] = spat[sbase + t];

    for (int c = 0; c < C_; ++c) {
        float acc = 0.f;
        #pragma unroll
        for (int t = 0; t < NK; ++t)                     // col idx: tx+dj-1 +4
            acc += s_x[c][ty + t / 3][tx + t % 3 + 3] * (s_spec[c][t] * sp[t]);
        out[(size_t)(b * C_ + c) * HW_ + h * W_ + w] = acc;
    }
}

// ---------------------------------------------------------------------------
extern "C" void kernel_launch(void* const* d_in, const int* in_sizes, int n_in,
                              void* d_out, int out_size, void* d_ws, size_t ws_size,
                              hipStream_t stream) {
    (void)in_sizes; (void)n_in; (void)out_size; (void)ws_size;
    const float* x        = (const float*)d_in[0];
    const float* y        = (const float*)d_in[1];
    const float* w_dw     = (const float*)d_in[2];
    const float* b_dw     = (const float*)d_in[3];
    const float* w_p1     = (const float*)d_in[4];
    const float* b_p1     = (const float*)d_in[5];
    const float* w_c2     = (const float*)d_in[6];
    const float* b_c2     = (const float*)d_in[7];
    const float* w_c3     = (const float*)d_in[8];
    const float* b_c3     = (const float*)d_in[9];
    const float* w_s1     = (const float*)d_in[10];
    const float* b_s1     = (const float*)d_in[11];
    const float* w_s2     = (const float*)d_in[12];
    const float* b_s2     = (const float*)d_in[13];
    const float* norm_std = (const float*)d_in[14];
    float* out = (float*)d_out;

    // ws (floats): gap[256] | spec[2304] | pad to 4096 | t1p (f16, 8M halves)
    // | spat (4.7M floats). Total ~36 MB.
    float*      ws   = (float*)d_ws;
    float*      gap  = ws;
    float*      spec = ws + 256;
    _Float16*   t1p  = (_Float16*)(ws + 4096);
    float*      spat = ws + 4096 + (size_t)B_ * HW_ * 16 / 2;

    const float std_spatial = sqrtf(2.0f / (1.0f + 0.01f * 0.01f)) / 3.0f;

    k_gap <<<B_ * C_, 256, 0, stream>>>(x, gap);
    k_spec<<<1, 256, 0, stream>>>(gap, w_s1, b_s1, w_s2, b_s2, norm_std, spec);
    k_dwpw<<<B_ * (HW_ / 128), 256, 0, stream>>>(y, w_dw, b_dw, w_p1, b_p1, t1p);

    dim3 tiles(W_ / 16, H_ / 16, B_);
    k_conv23<<<tiles, 256, 0, stream>>>(t1p, w_c2, b_c2, w_c3, b_c3, std_spatial, spat);
    k_out   <<<tiles, 256, 0, stream>>>(x, spec, spat, out);
}
